// RandomWalkPE_9672266350987
// MI455X (gfx1250) — 8-bit matrix operand order repaired and checked by reading, compiled, not run on hardware
//
#include <hip/hip_runtime.h>
#include <hip/hip_bf16.h>


typedef int v8i __attribute__((ext_vector_type(8)));

#define NN        4096
#define KDIM      128
#define ADJ_WORDS 128
#define PE_DIM    64
#define NUM_WALKS 10
#define WALK_LEN  5
#define NWALKS    (NN * NUM_WALKS)
#define D_OUT     192

#define ATILE_WORDS (16 * KDIM / 4)

__device__ __forceinline__ unsigned int hashu(unsigned int x) {
  x ^= x >> 16; x *= 0x7feb352dU;
  x ^= x >> 15; x *= 0x846ca68bU;
  x ^= x >> 16;
  return x;
}

__global__ void rw_zero_kernel(float* __restrict__ p, int n) {
  int i = blockIdx.x * blockDim.x + threadIdx.x;
  if (i < n) p[i] = 0.0f;
}

__global__ void rw_pack_kernel(const float* __restrict__ H,
                               unsigned char* __restrict__ Hu8) {
  int i = blockIdx.x * blockDim.x + threadIdx.x;
  if (i < NN * KDIM) Hu8[i] = (H[i] > 0.5f) ? (unsigned char)1 : (unsigned char)0;
}

__global__ __launch_bounds__(256)
void rw_adj_wmma_kernel(const unsigned char* __restrict__ Hu8,
                        unsigned int* __restrict__ adj) {
  __shared__ int aTileW[ATILE_WORDS];

  const int lane = threadIdx.x & 31;
  const int wave = threadIdx.x >> 5;
  const int r0   = blockIdx.x * 16;
  const int cw   = blockIdx.y * 256 + wave * 32;
  const int half = lane >> 4;
  const int ml   = lane & 15;

  if (threadIdx.x < 128) {
    const unsigned int lds_off = threadIdx.x * 16u;
    const unsigned char* gsrc  = Hu8 + (size_t)r0 * KDIM + threadIdx.x * 16;
    asm volatile("global_load_async_to_lds_b128 %0, %1, off"
                 :: "v"(lds_off), "v"(gsrc), "v"((int*)aTileW)
                 : "memory");
  }
  asm volatile("s_wait_asynccnt 0" ::: "memory");
  __syncthreads();

  const unsigned char* brow0 = Hu8 + (cw + ml)      * KDIM;
  const unsigned char* brow1 = Hu8 + (cw + 16 + ml) * KDIM;

  v8i c0 = {0, 0, 0, 0, 0, 0, 0, 0};
  v8i c1 = {0, 0, 0, 0, 0, 0, 0, 0};

#pragma unroll
  for (int kh = 0; kh < 2; ++kh) {
    const int k0 = kh * 64;
    v8i a, b0, b1;
#pragma unroll
    for (int i = 0; i < 8; ++i) {
      const int ka = k0 + ((i >> 2) * 32) + (((i >> 1) & 1) * 16) +
                     ((i & 1) * 4) + half * 8;
      a[i] = aTileW[ml * (KDIM / 4) + (ka >> 2)];
      const int kb = ka;
      b0[i] = *(const int*)(brow0 + kb);
      b1[i] = *(const int*)(brow1 + kb);
    }
    c0 = __builtin_amdgcn_wmma_i32_16x16x64_iu8(false, a, false, b0, c0, false, false);
    c1 = __builtin_amdgcn_wmma_i32_16x16x64_iu8(false, a, false, b1, c1, false, false);
  }

  const int word = cw >> 5;
#pragma unroll
  for (int j = 0; j < 8; ++j) {
    const int rA = r0 + j + 8 * half;
    const bool p0 = (c0[j] > 0) && (rA != (cw + ml));
    const bool p1 = (c1[j] > 0) && (rA != (cw + 16 + ml));
    const unsigned int b0m = __builtin_amdgcn_ballot_w32(p0);
    const unsigned int b1m = __builtin_amdgcn_ballot_w32(p1);
    if (lane == 0) {
      const unsigned int wlo = (b0m & 0xFFFFu) | ((b1m & 0xFFFFu) << 16);
      const unsigned int whi = (b0m >> 16) | (b1m & 0xFFFF0000u);
      adj[(r0 + j)     * ADJ_WORDS + word] = wlo;
      adj[(r0 + j + 8) * ADJ_WORDS + word] = whi;
    }
  }
}

__global__ void rw_degree_kernel(const unsigned int* __restrict__ adj,
                                 int* __restrict__ degree) {
  int r = blockIdx.x * blockDim.x + threadIdx.x;
  if (r >= NN) return;
  const unsigned long long* row = (const unsigned long long*)(adj + r * ADJ_WORDS);
  int s = 0;
#pragma unroll 4
  for (int w = 0; w < ADJ_WORDS / 2; ++w) s += __popcll(row[w]);
  degree[r] = s;
}

__global__ void rw_walk_kernel(const unsigned int* __restrict__ adj,
                               const int* __restrict__ degree,
                               int* __restrict__ wnodes,
                               int* __restrict__ wlen) {
  int w = blockIdx.x * blockDim.x + threadIdx.x;
  if (w >= NWALKS) return;
  int cur = w / NUM_WALKS;
  wnodes[w * WALK_LEN] = cur;
  int len = 1;
  for (int step = 1; step < WALK_LEN; ++step) {
    const int deg = degree[cur];
    if (deg <= 0) break;
    const unsigned long long* row =
        (const unsigned long long*)(adj + cur * ADJ_WORDS);
    __builtin_prefetch(row, 0, 1);
    unsigned int h = hashu((unsigned)(w * 8 + step) ^ 0x9E3779B9u);
    int r = (int)(h % (unsigned)deg);
    int nxt = cur;
    for (int wd = 0; wd < ADJ_WORDS / 2; ++wd) {
      unsigned long long word = row[wd];
      const int c = __popcll(word);
      if (r < c) {
        while (r-- > 0) word &= word - 1;
        nxt = wd * 64 + (int)(__ffsll((unsigned long long)word) - 1);
        break;
      }
      r -= c;
    }
    cur = nxt;
    wnodes[w * WALK_LEN + len] = cur;
    ++len;
  }
  wlen[w] = len;
}

__global__ __launch_bounds__(256)
void rw_scatter_kernel(const float* __restrict__ embed,
                       const int* __restrict__ wnodes,
                       const int* __restrict__ wlen,
                       float* __restrict__ pe_sum,
                       float* __restrict__ cnt) {
  const int w = blockIdx.x * 8 + (threadIdx.x >> 5);
  const int lane = threadIdx.x & 31;
  if (w >= NWALKS) return;
  const int len = wlen[w];

  int nodes[WALK_LEN];
#pragma unroll
  for (int p = 0; p < WALK_LEN; ++p) nodes[p] = wnodes[w * WALK_LEN + p];

  float a0 = 0.0f, a1 = 0.0f;
  for (int p = 0; p < len; ++p) {
    const float* e = embed + nodes[p] * PE_DIM;
    __builtin_prefetch(e, 0, 1);
    a0 += e[lane];
    a1 += e[32 + lane];
  }
  const float inv = 1.0f / (float)len;
  a0 *= inv;
  a1 *= inv;

  for (int p = 0; p < len; ++p) {
    float* ps = pe_sum + nodes[p] * PE_DIM;
    atomicAdd(ps + lane, a0);
    atomicAdd(ps + 32 + lane, a1);
    if (lane == 0) atomicAdd(cnt + nodes[p], 1.0f);
  }
}

__global__ void rw_finalize_kernel(const float* __restrict__ x,
                                   const float* __restrict__ pe_sum,
                                   const float* __restrict__ cnt,
                                   float* __restrict__ out) {
  int i = blockIdx.x * blockDim.x + threadIdx.x;
  if (i >= NN * D_OUT) return;
  const int n = i / D_OUT;
  const int c = i % D_OUT;
  if (c < 128) {
    out[i] = x[n * 128 + c];
  } else {
    const float k = cnt[n];
    out[i] = (k > 0.0f) ? pe_sum[n * PE_DIM + (c - 128)] / fmaxf(k, 1.0f) : 0.0f;
  }
}

extern "C" void kernel_launch(void* const* d_in, const int* in_sizes, int n_in,
                              void* d_out, int out_size, void* d_ws, size_t ws_size,
                              hipStream_t stream) {
  const float* x      = (const float*)d_in[0];
  const float* H      = (const float*)d_in[1];
  const float* embed  = (const float*)d_in[2];
  float* out          = (float*)d_out;

  char* ws = (char*)d_ws;
  unsigned char* Hu8    = (unsigned char*)(ws + 0);
  unsigned int*  adj    = (unsigned int*) (ws + 524288);
  int*           degree = (int*)          (ws + 2621440);
  int*           wnodes = (int*)          (ws + 2637824);
  int*           wlen   = (int*)          (ws + 3457024);
  float*         pe_sum = (float*)        (ws + 3620864);
  float*         cnt    = (float*)        (ws + 4669440);

  {
    const int n = NN * PE_DIM + NN;
    rw_zero_kernel<<<(n + 255) / 256, 256, 0, stream>>>(pe_sum, n);
  }
  {
    const int n = NN * KDIM;
    rw_pack_kernel<<<(n + 255) / 256, 256, 0, stream>>>(H, Hu8);
  }
  {
    dim3 grid(NN / 16, NN / 256);
    rw_adj_wmma_kernel<<<grid, 256, 0, stream>>>(Hu8, adj);
  }
  rw_degree_kernel<<<(NN + 255) / 256, 256, 0, stream>>>(adj, degree);
  rw_walk_kernel<<<(NWALKS + 255) / 256, 256, 0, stream>>>(adj, degree, wnodes, wlen);
  rw_scatter_kernel<<<NWALKS / 8, 256, 0, stream>>>(embed, wnodes, wlen, pe_sum, cnt);
  {
    const int n = NN * D_OUT;
    rw_finalize_kernel<<<(n + 255) / 256, 256, 0, stream>>>(x, pe_sum, cnt, out);
  }
}
